// AttentionLSTM_3238405341528
// MI455X (gfx1250) — matrix operand order repaired and checked by reading, not compiled, not run on hardware
//
#include <hip/hip_runtime.h>
#include <hip/hip_bf16.h>
#include <math.h>

// Problem constants (match reference): N=256, T=128, D=1024, H=1024
#define NB   256
#define TT   128
#define DD   1024
#define HH   1024
#define KTOT 3072   // D + H + H  (concatenated GEMM K)
#define FH   4096   // 4*H
#define KCH  512    // K-chunk staged in LDS
#define NCH  (KTOT / KCH)   // 6 chunks

typedef __attribute__((ext_vector_type(16))) __bf16 v16bf;
typedef __attribute__((ext_vector_type(8)))  __bf16 v8bf;
typedef __attribute__((ext_vector_type(8)))  float  v8f;
typedef __attribute__((ext_vector_type(4)))  int    v4i;

// gfx1250 async global->LDS copy path (ASYNCcnt), guarded so host pass and
// toolchains without the builtins still compile the synchronous fallback.
// Signature (from hipcc diagnostic): (AS1 v4i* gaddr, AS3 v4i* lds, imm, imm)
#if defined(__gfx1250__) && __has_builtin(__builtin_amdgcn_global_load_async_to_lds_b128) && __has_builtin(__builtin_amdgcn_s_wait_asynccnt)
#define USE_ASYNC 1
#define AS1 __attribute__((address_space(1)))
#define AS3 __attribute__((address_space(3)))
#else
#define USE_ASYNC 0
#endif

__device__ __forceinline__ float fast_sigmoid(float x) {
    return 1.f / (1.f + __expf(-x));
}
__device__ __forceinline__ float fast_tanh(float x) {
    // branchless; saturates correctly at +/-inf via expf overflow/underflow
    return 1.f - 2.f / (__expf(2.f * x) + 1.f);
}

// ---------------------------------------------------------------------------
// Prep 1: build WcatT (bf16, [4096 cols][3072 k]) = transpose of [Wx;Wh;Wattn]
// Column-of-B contiguous -> every WMMA B-fragment is one 32-byte per-lane load.
// ---------------------------------------------------------------------------
__global__ void alstm_wcat_transpose(const float* __restrict__ Wx,
                                     const float* __restrict__ Wh,
                                     const float* __restrict__ Wattn,
                                     __bf16* __restrict__ Wt) {
    size_t idx = (size_t)blockIdx.x * blockDim.x + threadIdx.x; // over 4096*3072
    int k   = (int)(idx % KTOT);
    int col = (int)(idx / KTOT);
    if (col >= FH) return;
    float v;
    if (k < 1024)       v = Wx[(size_t)k * FH + col];
    else if (k < 2048)  v = Wh[(size_t)(k - 1024) * FH + col];
    else                v = Wattn[(size_t)(k - 2048) * FH + col];
    Wt[(size_t)col * KTOT + k] = (__bf16)v;
}

// ---------------------------------------------------------------------------
// Prep 2: h0 = c0 = mean of A over the 4x4 grid
// ---------------------------------------------------------------------------
__global__ void alstm_init_state(const float* __restrict__ A,
                                 float* __restrict__ Hs,
                                 float* __restrict__ Cs) {
    int n = blockIdx.x;
    for (int j = threadIdx.x; j < HH; j += blockDim.x) {
        const float* a = A + ((size_t)n * HH + j) * 16;
        float s = 0.f;
#pragma unroll
        for (int l = 0; l < 16; ++l) s += a[l];
        float m = s * (1.0f / 16.0f);
        Hs[(size_t)n * HH + j] = m;
        Cs[(size_t)n * HH + j] = m;
    }
}

// ---------------------------------------------------------------------------
// Per-step kernel 1: 16-way spatial attention + pack
//   Z[n] = [ bf16(x[n,t,:]) | bf16(h[n,:]) | bf16(attn[n,:]) ]   (K = 3072)
// ---------------------------------------------------------------------------
__global__ void alstm_attn_pack(const float* __restrict__ x,
                                const float* __restrict__ A,
                                const float* __restrict__ Hs,
                                __bf16* __restrict__ Z,
                                int t) {
    __shared__ float wred[8][16];
    __shared__ float wts[16];

    const int n    = blockIdx.x;
    const int tid  = threadIdx.x;
    const int lane = tid & 31;
    const int wave = tid >> 5;

    const float* Arow = A  + (size_t)n * HH * 16;
    const float* hrow = Hs + (size_t)n * HH;

    float part[16];
#pragma unroll
    for (int l = 0; l < 16; ++l) part[l] = 0.f;
    for (int j = tid; j < HH; j += 256) {
        float hv = hrow[j];
        const float* a = Arow + (size_t)j * 16;
#pragma unroll
        for (int l = 0; l < 16; ++l) part[l] += hv * a[l];
    }
#pragma unroll
    for (int l = 0; l < 16; ++l) {
        float v = part[l];
        for (int off = 16; off > 0; off >>= 1) v += __shfl_xor(v, off, 32);
        if (lane == 0) wred[wave][l] = v;
    }
    __syncthreads();
    if (tid == 0) {
        const float scale = 0.03125f; // 1/sqrt(1024)
        float sc[16];
        float mx = -1e30f;
#pragma unroll
        for (int l = 0; l < 16; ++l) {
            float s = 0.f;
#pragma unroll
            for (int w = 0; w < 8; ++w) s += wred[w][l];
            s *= scale;
            sc[l] = s;
            mx = fmaxf(mx, s);
        }
        float sum = 0.f;
#pragma unroll
        for (int l = 0; l < 16; ++l) { sc[l] = __expf(sc[l] - mx); sum += sc[l]; }
        float inv = 1.f / sum;
#pragma unroll
        for (int l = 0; l < 16; ++l) wts[l] = sc[l] * inv;
    }
    __syncthreads();

    float wl[16];
#pragma unroll
    for (int l = 0; l < 16; ++l) wl[l] = wts[l];

    const float* xrow = x + ((size_t)n * TT + t) * DD;
    __bf16* zrow = Z + (size_t)n * KTOT;
    for (int j = tid; j < HH; j += 256) {
        const float* a = Arow + (size_t)j * 16;
        float acc = 0.f;
#pragma unroll
        for (int l = 0; l < 16; ++l) acc += a[l] * wl[l];
        zrow[j]        = (__bf16)xrow[j];   // xt
        zrow[1024 + j] = (__bf16)hrow[j];   // prev_h
        zrow[2048 + j] = (__bf16)acc;       // attn
    }
}

// ---------------------------------------------------------------------------
// Per-step kernel 2: fused GEMM + LSTM gates with LDS-staged A tiles.
//   pre = Z(256x3072) @ Wcat(3072x4096), gates applied in-wave.
// Grid: 128 blocks x 256 threads (= 1024 waves, ~1 wave/SIMD device-wide).
// Block -> 16-row m-tile; all 8 waves share the same A tile, so the Z tile is
// double-buffer staged global->LDS in K-chunks of 512 via async copies; each
// wave reads its A fragments with ds_load_b128 and streams its 4 gate-column
// B fragments from L2-resident bf16 weights.
// ---------------------------------------------------------------------------
__global__ void __launch_bounds__(256)
alstm_gemm_gates(const __bf16* __restrict__ Z,
                 const __bf16* __restrict__ Wt,
                 const float* __restrict__ bias,
                 float* __restrict__ Hs,
                 float* __restrict__ Cs,
                 float* __restrict__ out,
                 int t) {
    __shared__ __bf16 zs[2][16 * KCH];   // 2 x 16KB double buffer

    const int tid  = threadIdx.x;
    const int wave = tid >> 5;
    const int lane = tid & 31;
    const int half = lane >> 4;     // K-half select (A/B fragment layouts)
    const int lc   = lane & 15;

    const int mtile = blockIdx.x & 15;       // 16 tiles over N=256
    const int jgrp  = blockIdx.x >> 4;       // 8 groups of 8 j-tiles
    const int m0 = mtile * 16;
    const int j0 = (jgrp * 8 + wave) * 16;

    // ---- cooperative stage of one K-chunk of the Z m-tile into LDS --------
    // chunk = 16 rows x 512 bf16 = 1024 pieces of 16B; 4 pieces per thread.
    const __bf16* zg = Z + (size_t)m0 * KTOT;
    auto stage = [&](int c, int buf) {
        const __bf16* gbase = zg + c * KCH;
#pragma unroll
        for (int i = 0; i < 4; ++i) {
            int p    = tid + i * 256;
            int row  = p >> 6;
            int koff = (p & 63) << 3;
            const __bf16* src = gbase + (size_t)row * KTOT + koff;
            __bf16* dst = &zs[buf][row * KCH + koff];
#if USE_ASYNC
            __builtin_amdgcn_global_load_async_to_lds_b128(
                (AS1 v4i*)(void*)src, (AS3 v4i*)(void*)dst, 0, 0);
#else
            *(v8bf*)dst = *(const v8bf*)src;
#endif
        }
    };

    // B fragments: 32x16 bf16; lane lc holds column, half selects K 0..15 /
    // 16..31 -> 16 contiguous bf16 in WcatT ([col][k] layout).
    const __bf16* bp0 = Wt + (size_t)(0 * HH + j0 + lc) * KTOT + half * 16;
    const __bf16* bp1 = Wt + (size_t)(1 * HH + j0 + lc) * KTOT + half * 16;
    const __bf16* bp2 = Wt + (size_t)(2 * HH + j0 + lc) * KTOT + half * 16;
    const __bf16* bp3 = Wt + (size_t)(3 * HH + j0 + lc) * KTOT + half * 16;

    v8f acc0 = {0.f, 0.f, 0.f, 0.f, 0.f, 0.f, 0.f, 0.f};
    v8f acc1 = acc0, acc2 = acc0, acc3 = acc0;

    const int abase = half * 16;  // A-fragment K sub-offset (contiguous, matches B)

    stage(0, 0);

    for (int c = 0; c < NCH; ++c) {
        if (c + 1 < NCH) {
            stage(c + 1, (c + 1) & 1);
#if USE_ASYNC
            __builtin_amdgcn_s_wait_asynccnt(4);   // drain chunk c, keep c+1 in flight
#endif
        } else {
#if USE_ASYNC
            __builtin_amdgcn_s_wait_asynccnt(0);
#endif
        }
        __syncthreads();   // chunk c visible to all waves

        const __bf16* zbuf = &zs[c & 1][lc * KCH];
        const int kg = c * KCH;
#pragma unroll 2
        for (int kk = 0; kk < KCH; kk += 32) {
            v8bf alo = *(const v8bf*)(zbuf + kk + abase);        // ds_load_b128
            v8bf ahi = *(const v8bf*)(zbuf + kk + 8 + abase);    // ds_load_b128
            v16bf a;
#pragma unroll
            for (int i = 0; i < 8; ++i) { a[i] = alo[i]; a[8 + i] = ahi[i]; }

            v16bf b0 = *(const v16bf*)(bp0 + kg + kk);
            v16bf b1 = *(const v16bf*)(bp1 + kg + kk);
            v16bf b2 = *(const v16bf*)(bp2 + kg + kk);
            v16bf b3 = *(const v16bf*)(bp3 + kg + kk);

            acc0 = __builtin_amdgcn_wmma_f32_16x16x32_bf16(false, a, false, b0, (short)0, acc0, false, false);
            acc1 = __builtin_amdgcn_wmma_f32_16x16x32_bf16(false, a, false, b1, (short)0, acc1, false, false);
            acc2 = __builtin_amdgcn_wmma_f32_16x16x32_bf16(false, a, false, b2, (short)0, acc2, false, false);
            acc3 = __builtin_amdgcn_wmma_f32_16x16x32_bf16(false, a, false, b3, (short)0, acc3, false, false);
        }
        __syncthreads();   // all waves done reading before buffer reuse
    }

    // ---- in-wave LSTM gating: acc slot v holds ai/af/ao/ag of the same
    // (n, j) element (C/D layout: M = v + 8*half, Ncol = lc) -----------------
    const int jcol = j0 + lc;
    const float bi = bias[0 * HH + jcol];
    const float bf = bias[1 * HH + jcol];
    const float bo = bias[2 * HH + jcol];
    const float bg = bias[3 * HH + jcol];

#pragma unroll
    for (int v = 0; v < 8; ++v) {
        const int m = m0 + v + half * 8;          // batch index n
        const size_t sidx = (size_t)m * HH + jcol;
        float cprev = Cs[sidx];
        float ig = fast_sigmoid(acc0[v] + bi);
        float fg = fast_sigmoid(acc1[v] + bf);
        float og = fast_sigmoid(acc2[v] + bo);
        float gg = fast_tanh(acc3[v] + bg);
        float nc = fg * cprev + ig * gg;
        float nh = og * fast_tanh(nc);
        Cs[sidx] = nc;
        Hs[sidx] = nh;
        out[((size_t)m * TT + t) * HH + jcol] = nh;
    }
}

// ---------------------------------------------------------------------------
// Host-side launch
// ---------------------------------------------------------------------------
extern "C" void kernel_launch(void* const* d_in, const int* in_sizes, int n_in,
                              void* d_out, int out_size, void* d_ws, size_t ws_size,
                              hipStream_t stream) {
    (void)in_sizes; (void)n_in; (void)out_size; (void)ws_size;

    const float* x     = (const float*)d_in[0];  // (N,T,D)
    const float* A     = (const float*)d_in[1];  // (N,H,4,4)
    const float* Wx    = (const float*)d_in[2];  // (D,4H)
    const float* Wh    = (const float*)d_in[3];  // (H,4H)
    const float* Wattn = (const float*)d_in[4];  // (H,4H)
    const float* bias  = (const float*)d_in[5];  // (4H,)
    float* out = (float*)d_out;                  // (N,T,H)

    // Workspace carve (all chunks are multiples of 4KB -> aligned)
    char* p = (char*)d_ws;
    __bf16* Wt = (__bf16*)p; p += (size_t)FH * KTOT * sizeof(__bf16);   // ~24 MB
    __bf16* Z  = (__bf16*)p; p += (size_t)NB * KTOT * sizeof(__bf16);   // 1.5 MB
    float*  Hs = (float*)p;  p += (size_t)NB * HH * sizeof(float);      // 1 MB
    float*  Cs = (float*)p;  p += (size_t)NB * HH * sizeof(float);      // 1 MB

    // One-time prep (parallel, off the sequential chain)
    alstm_wcat_transpose<<<((size_t)FH * KTOT) / 256, 256, 0, stream>>>(Wx, Wh, Wattn, Wt);
    alstm_init_state<<<NB, 256, 0, stream>>>(A, Hs, Cs);

    // Sequential recurrence: 128 steps, 2 kernels each
    for (int t = 0; t < TT; ++t) {
        alstm_attn_pack<<<NB, 256, 0, stream>>>(x, A, Hs, Z, t);
        alstm_gemm_gates<<<128, 256, 0, stream>>>(Z, Wt, bias, Hs, Cs, out, t);
    }
}
